// CoverageLoss_3934190043474
// MI455X (gfx1250) — compile-verified, matrix operand order checked by reading, not run on hardware
//
#include <hip/hip_runtime.h>
#include <hip/hip_bf16.h>

#define T_LEN 4096
#define BINS  64
#define BLOCK 256

typedef __attribute__((ext_vector_type(16))) _Float16 v16h;
typedef __attribute__((ext_vector_type(8)))  float    v8f;
typedef __attribute__((ext_vector_type(4)))  float    v4f;
typedef __attribute__((ext_vector_type(4)))  int      v4i;

typedef __attribute__((address_space(1))) v4i* gas_v4i;
typedef __attribute__((address_space(3))) v4i* las_v4i;

#if __has_builtin(__builtin_amdgcn_global_load_async_to_lds_b128)
#define ASYNC_LDS 1
#else
#define ASYNC_LDS 0
#endif

#if __has_builtin(__builtin_amdgcn_s_wait_asynccnt)
#define ASYNC_WAIT(n) __builtin_amdgcn_s_wait_asynccnt(n)
#else
#define ASYNC_WAIT(n) asm volatile("s_wait_asynccnt %0" ::"i"(n) : "memory")
#endif

// histc binning per reference (f32 throughout)
static __device__ __forceinline__ void process_elems(
    const float* s_data, unsigned* s_occ, int ebase, int ln, int k0, int k1) {
  const float LOf  = -3.14159265f;                             // == -pi(f32)
  const float HIf  =  3.14159265f;                             // ==  pi(f32)
  const float wdtf = (float)((3.14159265 - (-3.14159265)) / 64.0);
#pragma unroll
  for (int k = k0; k < k1; ++k) {
    const int   e = ebase + ln + k * 32;     // element index within row
    const float x = s_data[2 * e + 0];
    const float y = s_data[2 * e + 1];
    const float a = atan2f(y, x);
    int idx = (int)floorf((a - LOf) / wdtf);
    idx = idx < 0 ? 0 : (idx > (BINS - 1) ? (BINS - 1) : idx);
    if (a >= LOf && a <= HIf) s_occ[idx] = 1u;  // idempotent store: race-free result
  }
}

// -------------------------------------------------------------------------
// Kernel 1: one workgroup (8 wave32) per row. Each wave async-stages its own
// 4 KB slice of the row into LDS and overlaps the second half of the DMA with
// atan2 compute on the first half (split s_wait_asynccnt). Occupancy flags in
// LDS; per-row loss = 1 - occupied/64.
// -------------------------------------------------------------------------
__global__ __launch_bounds__(BLOCK) void coverage_row_kernel(
    const float* __restrict__ cseq, float* __restrict__ row_loss) {
  __shared__ float    s_data[T_LEN * 2];   // 32 KB staged row
  __shared__ unsigned s_occ[BINS];

  const int t   = threadIdx.x;
  const int wv  = t >> 5;                  // wave id 0..7
  const int ln  = t & 31;                  // lane id
  const int row = blockIdx.x;
  const float* g = cseq + (size_t)row * (T_LEN * 2);

  const int ubase = wv * 256;              // this wave's region, 16B units
  const int ebase = wv * (T_LEN / 8);      // this wave's region, elements

#if ASYNC_LDS
  {
    gas_v4i gp = (gas_v4i)(float*)g;
    las_v4i lp = (las_v4i)s_data;
#pragma unroll
    for (int i = 0; i < 8; ++i) {          // 8 x b128 per lane, in-order completion
      const int off = ubase + i * 32 + ln;
      __builtin_amdgcn_global_load_async_to_lds_b128(gp + off, lp + off, 0, 0);
    }
  }
#else
#pragma unroll
  for (int i = 0; i < 8; ++i) {
    const int off = ubase + i * 32 + ln;
    ((v4f*)s_data)[off] = ((const v4f*)g)[off];
  }
#endif

  if (t < BINS) s_occ[t] = 0u;
  __syncthreads();                         // flags zeroed before any compute

#if ASYNC_LDS
  ASYNC_WAIT(4);                           // first 4 loads (first half) landed
#endif
  process_elems(s_data, s_occ, ebase, ln, 0, 8);   // overlap with tail DMA
#if ASYNC_LDS
  ASYNC_WAIT(0);                           // remaining loads landed
#endif
  process_elems(s_data, s_occ, ebase, ln, 8, 16);

  __syncthreads();
  if (t == 0) {
    int cnt = 0;
#pragma unroll
    for (int b = 0; b < BINS; ++b) cnt += (s_occ[b] != 0u) ? 1 : 0;
    row_loss[row] = 1.0f - (float)cnt * (1.0f / (float)BINS);
  }
}

// -------------------------------------------------------------------------
// Kernel 2: mean over per-row losses via WMMA.
// A = ones(16x32 f16) => C[i][j] += sum_k B[k][j]; every C row = column sums.
// Sum of lanes of c[0] = 2 * grand total. Row losses are multiples of 1/64
// (exact in f16, f32 accumulate) -> bit-exact, deterministic.
// Full 512-value chunks: unconditional vectorized loads (no exec juggling).
// -------------------------------------------------------------------------
__global__ __launch_bounds__(32) void mean_reduce_wmma(
    const float* __restrict__ vals, int n, float* __restrict__ out) {
  __shared__ float s_part[32];
  const int lane = threadIdx.x;            // single wave32, EXEC all ones

  v16h ones;
#pragma unroll
  for (int k = 0; k < 16; ++k) ones[k] = (_Float16)1.0f;

  v8f c = {};
  const int nfull = n >> 9;                // full 512-value chunks
  for (int i = 0; i < nfull; ++i) {
    const v4f* p = (const v4f*)(vals + i * 512 + lane * 16);
    const v4f a0 = p[0], a1 = p[1], a2 = p[2], a3 = p[3];
    v16h b;
#pragma unroll
    for (int k = 0; k < 4; ++k) {
      b[k + 0]  = (_Float16)a0[k];
      b[k + 4]  = (_Float16)a1[k];
      b[k + 8]  = (_Float16)a2[k];
      b[k + 12] = (_Float16)a3[k];
    }
    c = __builtin_amdgcn_wmma_f32_16x16x32_f16(false, ones, false, b,
                                               (short)0, c, false, false);
  }
  if (n & 511) {                           // uniform branch; skipped for n=2048
    v16h b;
#pragma unroll
    for (int k = 0; k < 16; ++k) {
      const int j = nfull * 512 + lane * 16 + k;
      b[k] = (j < n) ? (_Float16)vals[j] : (_Float16)0.0f;
    }
    c = __builtin_amdgcn_wmma_f32_16x16x32_f16(false, ones, false, b,
                                               (short)0, c, false, false);
  }

  s_part[lane] = c[0];
  __syncthreads();
  if (lane == 0) {
    float s = 0.0f;
#pragma unroll
    for (int l = 0; l < 32; ++l) s += s_part[l];   // fixed order: deterministic
    out[0] = s * (0.5f / (float)n);                // undo 2x duplication, / n
  }
}

// -------------------------------------------------------------------------
extern "C" void kernel_launch(void* const* d_in, const int* in_sizes, int n_in,
                              void* d_out, int out_size, void* d_ws, size_t ws_size,
                              hipStream_t stream) {
  (void)n_in; (void)out_size; (void)ws_size;
  const float* cseq     = (const float*)d_in[0];
  float*       out      = (float*)d_out;
  float*       row_loss = (float*)d_ws;            // rows floats of scratch

  const int rows = in_sizes[0] / (T_LEN * 2);      // 2048 for the reference shape

  coverage_row_kernel<<<rows, BLOCK, 0, stream>>>(cseq, row_loss);
  mean_reduce_wmma<<<1, 32, 0, stream>>>(row_loss, rows, out);
}
